// GMVAE_54305566491246
// MI455X (gfx1250) — matrix operand order repaired and checked by reading, not compiled, not run on hardware
//
#include <hip/hip_runtime.h>
#include <hip/hip_bf16.h>

typedef __bf16 bf16_t;
typedef __attribute__((ext_vector_type(16))) __bf16 v16bf;
typedef __attribute__((ext_vector_type(8)))  float  v8f;
typedef __attribute__((ext_vector_type(4)))  float  v4f;

static constexpr int kB = 8192;   // batch
static constexpr int kX = 784;    // x dim
static constexpr int kZ = 10;     // latent dim
static constexpr int kK = 2000;   // mixture components
static constexpr int kH = 300;    // hidden

static constexpr int kXp = 800;   // 784 -> pad to 25*32
static constexpr int kHp = 320;   // 300 -> pad to 10*32
static constexpr int kEp = 32;    // 20  -> pad to 32
static constexpr int kZp = 32;    // 10  -> pad to 32

static constexpr int kMaxKt = 25; // largest K-tile count (kXp/32)

#define LOG2PI 1.8378770664093453f
#define LOGK   7.6009024595420815f   // ln(2000)

__device__ __forceinline__ float softplus_f(float x) {
  return fmaxf(x, 0.0f) + log1pf(__expf(-fabsf(x)));
}

// ---------------------------------------------------------------- utilities
__global__ void fill_zero_f32(float* __restrict__ p, int n) {
  int i = blockIdx.x * blockDim.x + threadIdx.x;
  if (i < n) p[i] = 0.0f;
}

// pad+convert row-major f32 (rows x Kd) into bf16 (rows x Kpad), zero-filled
__global__ void pad_rows_bf16(const float* __restrict__ in, bf16_t* __restrict__ out,
                              int rows, int Kd, int Kpad) {
  int i = blockIdx.x * blockDim.x + threadIdx.x;
  if (i >= rows * Kpad) return;
  int r = i / Kpad, k = i - r * Kpad;
  out[i] = (k < Kd) ? (bf16_t)in[(size_t)r * Kd + k] : (bf16_t)0.0f;
}

// Pack W (Kd x N row-major f32) into WMMA-B fragment order, zero padded.
// Tile (nt,kt): 32 lanes x 16 contiguous bf16 at ((nt*nKt + kt)*32 + lane)*16.
// Lane: col = nt*16 + (lane&15); element j -> k = kt*32 + (lane>>4)*16 + j.
// For fixed nt the whole k-column (nKt*512 elems) is contiguous.
__global__ void pack_weight_bf16(const float* __restrict__ W, bf16_t* __restrict__ Wp,
                                 int Kd, int N, int nKt, int nNt) {
  int idx = blockIdx.x * blockDim.x + threadIdx.x;
  if (idx >= nNt * nKt * 32) return;
  int lane = idx & 31;
  int t    = idx >> 5;
  int kt   = t % nKt;
  int nt   = t / nKt;
  int n    = nt * 16 + (lane & 15);
  int kb   = kt * 32 + (lane >> 4) * 16;
  bf16_t* dst = Wp + (size_t)idx * 16;
#pragma unroll
  for (int j = 0; j < 16; ++j) {
    int k = kb + j;
    dst[j] = (k < Kd && n < N) ? (bf16_t)W[(size_t)k * N + n] : (bf16_t)0.0f;
  }
}

// ------------------------------------------------------------- WMMA GEMM
// A: kB x Kpad bf16 (padded, row-major). Wp: packed fragments.
// All 8 waves of a block share the same two B n-tiles -> stage both packed
// B tile-columns in LDS once per block; k-loop reads B via ds_load_b128 and
// streams A from global (double-buffered). Block: 128 x 32 tile, exact grids,
// no early returns (barrier), EXEC all-ones at every WMMA.
// EPI 0: outB[row*Npad+col] = col<N ? elu(acc+bias) : 0   (bf16)
// EPI 1: outF[row*Npad+col] = col<N ? acc+bias : 0        (f32)
// EPI 2: t = x*logit - softplus(logit); row-sum -> atomicAdd(rowacc[row])
template <int EPI>
__global__ void __launch_bounds__(256)
gemm_bf16_wmma(const bf16_t* __restrict__ A, const bf16_t* __restrict__ Wp,
               const float* __restrict__ bias, bf16_t* __restrict__ outB,
               float* __restrict__ outF, const float* __restrict__ Xf,
               float* __restrict__ rowacc, int N, int Npad, int Kpad, int nKt) {
  __shared__ __align__(16) bf16_t sB[2 * kMaxKt * 512];   // 51.2 KB max

  int lane = threadIdx.x & 31;
  int w    = threadIdx.x >> 5;
  int tile_m = blockIdx.x * 128 + w * 16;
  int nt0    = blockIdx.y * 2;                 // two 16-wide n tiles per block

  // cooperative stage of both contiguous B tile-columns into LDS
  {
    const v4f* s0 = (const v4f*)(Wp + (size_t)nt0 * nKt * 512);
    const v4f* s1 = (const v4f*)(Wp + (size_t)(nt0 + 1) * nKt * 512);
    v4f* dst = (v4f*)sB;
    int chunks = nKt * 64;                     // 16B chunks per tile-column
    for (int i = threadIdx.x; i < chunks; i += 256) {
      dst[i]          = s0[i];
      dst[chunks + i] = s1[i];
    }
  }
  __syncthreads();

  const bf16_t* ap = A + (size_t)(tile_m + (lane & 15)) * Kpad + (lane >> 4) * 16;
  const v4f* lb = (const v4f*)sB + lane * 2;   // per-lane base, 32B per kt-block
  const int  cs = nKt * 64;                    // v4f stride between tile columns

  union Frag { v16bf v; v4f q[2]; };
  Frag ac, an, b0, b1;

  ac.q[0] = *(const v4f*)(ap);
  ac.q[1] = *(const v4f*)(ap + 8);

  v8f acc0 = {}, acc1 = {};
  for (int kt = 0; kt < nKt - 1; ++kt) {
    ap += 32;
    an.q[0] = *(const v4f*)(ap);               // next A stays in flight
    an.q[1] = *(const v4f*)(ap + 8);
    b0.q[0] = lb[kt * 64];                     // B from LDS (ds_load_b128)
    b0.q[1] = lb[kt * 64 + 1];
    b1.q[0] = lb[cs + kt * 64];
    b1.q[1] = lb[cs + kt * 64 + 1];
    acc0 = __builtin_amdgcn_wmma_f32_16x16x32_bf16(false, ac.v, false, b0.v,
                                                   (short)0, acc0, false, false);
    acc1 = __builtin_amdgcn_wmma_f32_16x16x32_bf16(false, ac.v, false, b1.v,
                                                   (short)0, acc1, false, false);
    ac = an;
  }
  b0.q[0] = lb[(nKt - 1) * 64];
  b0.q[1] = lb[(nKt - 1) * 64 + 1];
  b1.q[0] = lb[cs + (nKt - 1) * 64];
  b1.q[1] = lb[cs + (nKt - 1) * 64 + 1];
  acc0 = __builtin_amdgcn_wmma_f32_16x16x32_bf16(false, ac.v, false, b0.v,
                                                 (short)0, acc0, false, false);
  acc1 = __builtin_amdgcn_wmma_f32_16x16x32_bf16(false, ac.v, false, b1.v,
                                                 (short)0, acc1, false, false);

  int col0  = nt0 * 16 + (lane & 15);
  int col1  = col0 + 16;
  int rbase = tile_m + (lane >> 4) * 8;
  bool cok0 = (col0 < N), cok1 = (col1 < N);
  float bv0 = bias[min(col0, N - 1)];
  float bv1 = bias[min(col1, N - 1)];

#pragma unroll
  for (int r = 0; r < 8; ++r) {
    int row  = rbase + r;
    float v0 = acc0[r] + bv0;
    float v1 = acc1[r] + bv1;
    if (EPI == 0) {
      float e0 = (v0 > 0.0f) ? v0 : (__expf(v0) - 1.0f);
      float e1 = (v1 > 0.0f) ? v1 : (__expf(v1) - 1.0f);
      outB[(size_t)row * Npad + col0] = cok0 ? (bf16_t)e0 : (bf16_t)0.0f;
      outB[(size_t)row * Npad + col1] = cok1 ? (bf16_t)e1 : (bf16_t)0.0f;
    } else if (EPI == 1) {
      outF[(size_t)row * Npad + col0] = cok0 ? v0 : 0.0f;
      outF[(size_t)row * Npad + col1] = cok1 ? v1 : 0.0f;
    } else {
      float x0 = Xf[(size_t)row * N + min(col0, N - 1)];
      float x1 = Xf[(size_t)row * N + min(col1, N - 1)];
      float t0 = cok0 ? (x0 * v0 - (fmaxf(v0, 0.0f) + log1pf(__expf(-fabsf(v0))))) : 0.0f;
      float t1 = cok1 ? (x1 * v1 - (fmaxf(v1, 0.0f) + log1pf(__expf(-fabsf(v1))))) : 0.0f;
      float t = t0 + t1;                       // same row, different cols
#pragma unroll
      for (int off = 1; off < 16; off <<= 1) t += __shfl_xor(t, off, 32);
      if ((lane & 15) == 0) atomicAdd(&rowacc[row], t);
    }
  }
}

// --------------------------------------------- reparameterized sample + log q
__global__ void z_sample_kernel(const float* __restrict__ enc, const float* __restrict__ eps,
                                float* __restrict__ zf, bf16_t* __restrict__ zbp,
                                float* __restrict__ logqz) {
  int b = blockIdx.x * blockDim.x + threadIdx.x;
  if (b >= kB) return;
  float lq = 0.0f;
#pragma unroll
  for (int j = 0; j < kZ; ++j) {
    float qm = enc[b * kEp + j];
    float qh = enc[b * kEp + kZ + j];
    float qv = softplus_f(qh) + 1e-8f;
    float e  = eps[b * kZ + j];
    float z  = qm + sqrtf(qv) * e;
    zf[b * kZ + j]  = z;
    zbp[b * kZp + j] = (bf16_t)z;
    lq += -0.5f * (LOG2PI + __logf(qv) + e * e);   // (z-qm)^2/qv == e^2
  }
#pragma unroll
  for (int j = kZ; j < kZp; ++j) zbp[b * kZp + j] = (bf16_t)0.0f;
  logqz[b] = lq;
}

// ---------------------------------------------------- GMM prior coefficients
// comp[b,k] = A_k + sum_j z_j*(m/v)_kj + z_j^2*(-0.5/v)_kj
__global__ void prep_prior_kernel(const float* __restrict__ zpre, float* __restrict__ prior) {
  int k = blockIdx.x * blockDim.x + threadIdx.x;
  if (k >= kK) return;
  float A = 0.0f;
#pragma unroll
  for (int j = 0; j < kZ; ++j) {
    float m = zpre[(size_t)k * kZ + j];
    float h = zpre[(size_t)(kK + k) * kZ + j];
    float v = softplus_f(h) + 1e-8f;
    prior[k * 21 + j]      = m / v;
    prior[k * 21 + kZ + j] = -0.5f / v;
    A += -0.5f * (LOG2PI + __logf(v) + m * m / v);
  }
  prior[k * 21 + 2 * kZ] = A;
}

// ---------------------------------------------------- online logsumexp over K
#define GMM_CH 128
__global__ void __launch_bounds__(256)
gmm_logpz_kernel(const float* __restrict__ zf, const float* __restrict__ prior,
                 float* __restrict__ logpz) {
  __shared__ float sp[GMM_CH * 21];
  int b = blockIdx.x * blockDim.x + threadIdx.x;   // kB is a multiple of 256
  float zr[kZ], z2[kZ];
#pragma unroll
  for (int j = 0; j < kZ; ++j) { float z = zf[b * kZ + j]; zr[j] = z; z2[j] = z * z; }
  float mx = -3.0e38f, ss = 0.0f;
  for (int k0 = 0; k0 < kK; k0 += GMM_CH) {
    int cnt = min(GMM_CH, kK - k0);
    __syncthreads();
    for (int i = threadIdx.x; i < cnt * 21; i += blockDim.x) sp[i] = prior[k0 * 21 + i];
    __syncthreads();
    for (int kk = 0; kk < cnt; ++kk) {
      const float* p = &sp[kk * 21];
      float c = p[2 * kZ];
#pragma unroll
      for (int j = 0; j < kZ; ++j) c += zr[j] * p[j] + z2[j] * p[kZ + j];
      if (c > mx) { ss = ss * __expf(mx - c) + 1.0f; mx = c; }
      else        { ss += __expf(c - mx); }
    }
  }
  logpz[b] = mx + __logf(ss) - LOGK;
}

// ----------------------------------------------------------- final reduction
__global__ void finalize_kernel(const float* __restrict__ logpx, const float* __restrict__ logqz,
                                const float* __restrict__ logpz, float* __restrict__ out) {
  __shared__ float s1[256], s2[256];
  int t = threadIdx.x;
  float a1 = 0.0f, a2 = 0.0f;
  for (int b = t; b < kB; b += 256) {
    a1 += -logpx[b];
    a2 += logqz[b] - logpz[b];
  }
  s1[t] = a1; s2[t] = a2;
  __syncthreads();
  for (int s = 128; s > 0; s >>= 1) {
    if (t < s) { s1[t] += s1[t + s]; s2[t] += s2[t + s]; }
    __syncthreads();
  }
  if (t == 0) {
    float rec = s1[0] / (float)kB;
    float kl  = s2[0] / (float)kB;
    out[0] = rec + kl;   // nelbo
    out[1] = kl;
    out[2] = rec;
  }
}

// ------------------------------------------------------------------- driver
extern "C" void kernel_launch(void* const* d_in, const int* in_sizes, int n_in,
                              void* d_out, int out_size, void* d_ws, size_t ws_size,
                              hipStream_t stream) {
  const float* x    = (const float*)d_in[0];
  const float* eps  = (const float*)d_in[1];
  const float* zpre = (const float*)d_in[2];
  const float* ew1  = (const float*)d_in[3];
  const float* eb1  = (const float*)d_in[4];
  const float* ew2  = (const float*)d_in[5];
  const float* eb2  = (const float*)d_in[6];
  const float* ew3  = (const float*)d_in[7];
  const float* eb3  = (const float*)d_in[8];
  const float* dw1  = (const float*)d_in[9];
  const float* db1  = (const float*)d_in[10];
  const float* dw2  = (const float*)d_in[11];
  const float* db2  = (const float*)d_in[12];
  const float* dw3  = (const float*)d_in[13];
  const float* db3  = (const float*)d_in[14];
  float* out = (float*)d_out;

  char* base = (char*)d_ws;
  size_t off = 0;
  auto alloc = [&](size_t bytes) -> char* {
    char* p = base + off;
    off = (off + bytes + 255) & ~(size_t)255;
    return p;
  };

  // tile counts per weight
  const int nKt_x = kXp / 32, nKt_h = kHp / 32, nKt_z = kZp / 32;
  const int nNt_h = kHp / 16, nNt_e = kEp / 16, nNt_x = kXp / 16;

  bf16_t* xpb   = (bf16_t*)alloc((size_t)kB * kXp * 2);
  bf16_t* ew1p  = (bf16_t*)alloc((size_t)nNt_h * nKt_x * 512 * 2);
  bf16_t* ew2p  = (bf16_t*)alloc((size_t)nNt_h * nKt_h * 512 * 2);
  bf16_t* ew3p  = (bf16_t*)alloc((size_t)nNt_e * nKt_h * 512 * 2);
  bf16_t* dw1p  = (bf16_t*)alloc((size_t)nNt_h * nKt_z * 512 * 2);
  bf16_t* dw2p  = (bf16_t*)alloc((size_t)nNt_h * nKt_h * 512 * 2);
  bf16_t* dw3p  = (bf16_t*)alloc((size_t)nNt_x * nKt_h * 512 * 2);
  bf16_t* h1b   = (bf16_t*)alloc((size_t)kB * kHp * 2);
  bf16_t* h2b   = (bf16_t*)alloc((size_t)kB * kHp * 2);
  bf16_t* d1b   = (bf16_t*)alloc((size_t)kB * kHp * 2);
  bf16_t* d2b   = (bf16_t*)alloc((size_t)kB * kHp * 2);
  float*  enc   = (float*)alloc((size_t)kB * kEp * 4);
  float*  zf    = (float*)alloc((size_t)kB * kZ * 4);
  bf16_t* zbp   = (bf16_t*)alloc((size_t)kB * kZp * 2);
  float*  lqz   = (float*)alloc((size_t)kB * 4);
  float*  lpz   = (float*)alloc((size_t)kB * 4);
  float*  lpx   = (float*)alloc((size_t)kB * 4);
  float*  prior = (float*)alloc((size_t)kK * 21 * 4);

  const int T = 256;
  auto cgrid = [](int n) { return (n + 255) / 256; };

  // data prep: pad x, pack weights into B-fragment order
  pad_rows_bf16<<<cgrid(kB * kXp), T, 0, stream>>>(x, xpb, kB, kX, kXp);
  pack_weight_bf16<<<cgrid(nNt_h * nKt_x * 32), T, 0, stream>>>(ew1, ew1p, kX, kH, nKt_x, nNt_h);
  pack_weight_bf16<<<cgrid(nNt_h * nKt_h * 32), T, 0, stream>>>(ew2, ew2p, kH, kH, nKt_h, nNt_h);
  pack_weight_bf16<<<cgrid(nNt_e * nKt_h * 32), T, 0, stream>>>(ew3, ew3p, kH, 2 * kZ, nKt_h, nNt_e);
  pack_weight_bf16<<<cgrid(nNt_h * nKt_z * 32), T, 0, stream>>>(dw1, dw1p, kZ, kH, nKt_z, nNt_h);
  pack_weight_bf16<<<cgrid(nNt_h * nKt_h * 32), T, 0, stream>>>(dw2, dw2p, kH, kH, nKt_h, nNt_h);
  pack_weight_bf16<<<cgrid(nNt_x * nKt_h * 32), T, 0, stream>>>(dw3, dw3p, kH, kX, nKt_h, nNt_x);

  dim3 blk(256);
  dim3 gH(kB / 128, kHp / 32);   // N=300 (Npad 320)
  dim3 gE(kB / 128, kEp / 32);   // N=20  (Npad 32)
  dim3 gX(kB / 128, kXp / 32);   // N=784 (Npad 800)

  // encoder
  gemm_bf16_wmma<0><<<gH, blk, 0, stream>>>(xpb, ew1p, eb1, h1b, nullptr, nullptr, nullptr, kH, kHp, kXp, nKt_x);
  gemm_bf16_wmma<0><<<gH, blk, 0, stream>>>(h1b, ew2p, eb2, h2b, nullptr, nullptr, nullptr, kH, kHp, kHp, nKt_h);
  gemm_bf16_wmma<1><<<gE, blk, 0, stream>>>(h2b, ew3p, eb3, nullptr, enc, nullptr, nullptr, 2 * kZ, kEp, kHp, nKt_h);

  // sample z, log q(z|x)
  z_sample_kernel<<<kB / 256, T, 0, stream>>>(enc, eps, zf, zbp, lqz);

  // decoder
  gemm_bf16_wmma<0><<<gH, blk, 0, stream>>>(zbp, dw1p, db1, d1b, nullptr, nullptr, nullptr, kH, kHp, kZp, nKt_z);
  gemm_bf16_wmma<0><<<gH, blk, 0, stream>>>(d1b, dw2p, db2, d2b, nullptr, nullptr, nullptr, kH, kHp, kHp, nKt_h);

  // fused Bernoulli log-likelihood epilogue (logits never materialized)
  fill_zero_f32<<<cgrid(kB), T, 0, stream>>>(lpx, kB);
  gemm_bf16_wmma<2><<<gX, blk, 0, stream>>>(d2b, dw3p, db3, nullptr, nullptr, x, lpx, kX, kXp, kHp, nKt_h);

  // GMM prior: log p(z) via online logsumexp
  prep_prior_kernel<<<cgrid(kK), T, 0, stream>>>(zpre, prior);
  gmm_logpz_kernel<<<kB / 256, T, 0, stream>>>(zf, prior, lpz);

  // final scalars
  finalize_kernel<<<1, T, 0, stream>>>(lpx, lqz, lpz, out);
}
